// MultiheadAttention_67095979098402
// MI455X (gfx1250) — matrix operand order repaired and checked by reading, not compiled, not run on hardware
//
#include <hip/hip_runtime.h>

// ---------------- CDNA5 WMMA plumbing (gfx1250, wave32) ----------------
typedef __attribute__((ext_vector_type(16))) __bf16 v16bf;
typedef __attribute__((ext_vector_type(8)))  float  v8f;
typedef __attribute__((ext_vector_type(4)))  unsigned int u32x4;

#define B_   2
#define S_   4096
#define H_   512
#define NH_  8
#define DK_  64
#define BS_  (B_ * S_)   // 8192 rows total

union FragU { v16bf v; u32x4 u[2]; };

// A-matrix fragment (16x32 bf16): lane holds row m=lane%16.
// K offsets: {half*8 + 0..7, 16 + half*8 + 0..7}. Pass p = &src[row*ld + kbase + half*8].
__device__ __forceinline__ v16bf frag_a(const __bf16* p) {
  FragU f;
  f.u[0] = *(const u32x4*)(p);
  f.u[1] = *(const u32x4*)(p + 16);
  return f.v;
}
// B-matrix fragment (32x16 bf16): lane holds col n=lane%16.
// K offsets: {half*8 + 0..7, 16 + half*8 + 0..7}. Pass p = &srcT[col*ld + kbase + half*8].
__device__ __forceinline__ v16bf frag_b(const __bf16* p) {
  FragU f;
  f.u[0] = *(const u32x4*)(p);
  f.u[1] = *(const u32x4*)(p + 16);
  return f.v;
}
__device__ __forceinline__ v8f wmma_bf16(v16bf a, v16bf b, v8f c) {
  // (neg_a, A, neg_b, B, c_mod, C, reuse_a, reuse_b)
  return __builtin_amdgcn_wmma_f32_16x16x32_bf16(false, a, false, b, (short)0, c,
                                                 false, false);
}
__device__ __forceinline__ v8f vzero8() {
  v8f z;
#pragma unroll
  for (int i = 0; i < 8; ++i) z[i] = 0.0f;
  return z;
}

// ---------------- f32 -> bf16 convert (vectorized) ----------------
__global__ __launch_bounds__(256) void mha_cvt_bf16(const float* __restrict__ x,
                                                    __bf16* __restrict__ y, int n) {
  int i = (blockIdx.x * 256 + threadIdx.x) * 4;
  if (i < n) {
    float4 f = *(const float4*)(x + i);
    y[i + 0] = (__bf16)f.x;
    y[i + 1] = (__bf16)f.y;
    y[i + 2] = (__bf16)f.z;
    y[i + 3] = (__bf16)f.w;
  }
}

// ---------------- weight transpose + convert: WT[n][k] = W[k][n] ----------------
__global__ __launch_bounds__(256) void mha_wt_bf16(const float* __restrict__ Wsrc,
                                                   __bf16* __restrict__ WT) {
  int idx = blockIdx.x * 256 + threadIdx.x;  // idx = n*512 + k
  int n = idx >> 9;
  int k = idx & 511;
  WT[idx] = (__bf16)Wsrc[k * H_ + n];
}

// ---------------- GEMM: C = alpha * (A[MxK] * (BT[NxK])^T + bias) ----------------
// Block: 256 thr (8 waves), tile 128(M) x 128(N). Wave = 32x64 = 2x4 WMMA tiles.
// OUTMODE 0: bf16 row-major [M][N]; 1: bf16 transposed [N][M] (for V); 2: f32 [M][N].
template <int OUTMODE>
__global__ __launch_bounds__(256) void mha_gemm(const __bf16* __restrict__ A,
                                                const __bf16* __restrict__ BT,
                                                const float* __restrict__ bias,
                                                void* __restrict__ Cout,
                                                int M, int N, int K, float alpha) {
  const int lane = threadIdx.x & 31;
  const int wave = threadIdx.x >> 5;
  const int hf   = lane >> 4;
  const int lr   = lane & 15;
  const int m0 = blockIdx.x * 128 + (wave >> 1) * 32;
  const int n0 = blockIdx.y * 128 + (wave & 1) * 64;

  v8f acc[2][4];
#pragma unroll
  for (int i = 0; i < 2; ++i)
#pragma unroll
    for (int j = 0; j < 4; ++j) acc[i][j] = vzero8();

  for (int kk = 0; kk < K; kk += 32) {
    v16bf af[2], bf[4];
#pragma unroll
    for (int i = 0; i < 2; ++i)
      af[i] = frag_a(A + (size_t)(m0 + i * 16 + lr) * K + kk + hf * 8);
#pragma unroll
    for (int j = 0; j < 4; ++j)
      bf[j] = frag_b(BT + (size_t)(n0 + j * 16 + lr) * K + kk + hf * 8);
#pragma unroll
    for (int i = 0; i < 2; ++i)
#pragma unroll
      for (int j = 0; j < 4; ++j) acc[i][j] = wmma_bf16(af[i], bf[j], acc[i][j]);
  }

  // C/D layout: lane holds n = lane%16, rows m = hf*8 + r (r = acc element)
#pragma unroll
  for (int i = 0; i < 2; ++i)
#pragma unroll
    for (int j = 0; j < 4; ++j) {
      const int nn = n0 + j * 16 + lr;
      const float bb = bias[nn];
#pragma unroll
      for (int r = 0; r < 8; ++r) {
        const int mm = m0 + i * 16 + hf * 8 + r;
        float v = (acc[i][j][r] + bb) * alpha;
        if (OUTMODE == 0)      ((__bf16*)Cout)[(size_t)mm * N + nn] = (__bf16)v;
        else if (OUTMODE == 1) ((__bf16*)Cout)[(size_t)nn * M + mm] = (__bf16)v;
        else                   ((float*)Cout)[(size_t)mm * N + nn] = v;
      }
    }
}

// ---------------- Flash attention ----------------
// Grid: (S/64, NH, B); block 128 (4 waves). Wave owns 16 q-rows, streams keys
// in 64-wide tiles. Q (pre-scaled by 1/sqrt(dk)) and K in [s][dk] bf16,
// V transposed [dk][s] bf16. Row-sum of P is computed on the matrix core:
// rowsum = P @ ones(64x16), whose C-layout matches lsum[] register mapping.
__global__ __launch_bounds__(128) void mha_flash(const __bf16* __restrict__ Qb,
                                                 const __bf16* __restrict__ Kb,
                                                 const __bf16* __restrict__ Vt,
                                                 __bf16* __restrict__ Ob) {
  const int lane = threadIdx.x & 31;
  const int wave = threadIdx.x >> 5;
  const int hf = lane >> 4;
  const int lr = lane & 15;
  const int b = blockIdx.z;
  const int h = blockIdx.y;
  const int q0 = blockIdx.x * 64 + wave * 16;

  __shared__ __align__(16) __bf16 Plds[4][16][72];  // 72 = 64 + pad, keeps 16B align

  // constant ones B-fragment for row-sum WMMA
  v16bf onesb;
#pragma unroll
  for (int i = 0; i < 16; ++i) onesb[i] = (__bf16)1.0f;

  // Q fragments: 2 K-steps covering dk = 0..63 (Q already includes 1/sqrt(dk))
  v16bf qf[2];
  {
    const __bf16* qp = Qb + (size_t)(b * S_ + q0 + lr) * H_ + h * DK_;
#pragma unroll
    for (int ks = 0; ks < 2; ++ks) qf[ks] = frag_a(qp + ks * 32 + hf * 8);
  }

  float mrow[8], lsum[8];
  v8f oacc[4];
#pragma unroll
  for (int r = 0; r < 8; ++r) { mrow[r] = -3.0e30f; lsum[r] = 0.0f; }
#pragma unroll
  for (int t = 0; t < 4; ++t) oacc[t] = vzero8();

  for (int kv = 0; kv < S_; kv += 64) {
    // ---- S = Q @ K^T (16 x 64) ----
    v8f sacc[4];
#pragma unroll
    for (int j = 0; j < 4; ++j) sacc[j] = vzero8();
#pragma unroll
    for (int j = 0; j < 4; ++j) {
      const __bf16* kp = Kb + (size_t)(b * S_ + kv + j * 16 + lr) * H_ + h * DK_;
#pragma unroll
      for (int ks = 0; ks < 2; ++ks) {
        v16bf kf = frag_b(kp + ks * 32 + hf * 8);
        sacc[j] = wmma_bf16(qf[ks], kf, sacc[j]);
      }
    }
    if (kv + 64 < S_) {  // prefetch next K/V tiles -> global_prefetch
      __builtin_prefetch(Kb + (size_t)(b * S_ + kv + 64 + lr) * H_ + h * DK_, 0, 0);
      __builtin_prefetch(Vt + (size_t)(h * DK_ + lr) * BS_ + b * S_ + kv + 64, 0, 0);
    }

    // ---- online softmax; C-layout row m = hf*8 + r lives in one 16-lane half ----
#pragma unroll
    for (int r = 0; r < 8; ++r) {
      float s0 = sacc[0][r], s1 = sacc[1][r], s2 = sacc[2][r], s3 = sacc[3][r];
      float mx = fmaxf(fmaxf(s0, s1), fmaxf(s2, s3));
#pragma unroll
      for (int mk = 1; mk < 16; mk <<= 1) mx = fmaxf(mx, __shfl_xor(mx, mk, 32));
      const float mnew = fmaxf(mrow[r], mx);
      const float corr = __expf(mrow[r] - mnew);
      mrow[r] = mnew;
      float p0 = __expf(s0 - mnew), p1 = __expf(s1 - mnew);
      float p2 = __expf(s2 - mnew), p3 = __expf(s3 - mnew);
      __bf16* prow = &Plds[wave][hf * 8 + r][lr];
      prow[0]  = (__bf16)p0;
      prow[16] = (__bf16)p1;
      prow[32] = (__bf16)p2;
      prow[48] = (__bf16)p3;
      lsum[r] *= corr;
#pragma unroll
      for (int t = 0; t < 4; ++t) oacc[t][r] *= corr;
    }
    asm volatile("s_wait_dscnt 0" ::: "memory");  // P tile visible before re-read

    // ---- reload P as A-fragments; row-sum and O-update on matrix core ----
    v16bf pf[2];
#pragma unroll
    for (int ks = 0; ks < 2; ++ks)
      pf[ks] = frag_a(&Plds[wave][lr][ks * 32 + hf * 8]);

    v8f lacc = vzero8();  // lacc[r] = rowsum(P) for row hf*8 + r
#pragma unroll
    for (int ks = 0; ks < 2; ++ks) lacc = wmma_bf16(pf[ks], onesb, lacc);
#pragma unroll
    for (int r = 0; r < 8; ++r) lsum[r] += lacc[r];

#pragma unroll
    for (int t = 0; t < 4; ++t) {
      const __bf16* vp = Vt + (size_t)(h * DK_ + t * 16 + lr) * BS_ + b * S_ + kv;
#pragma unroll
      for (int ks = 0; ks < 2; ++ks) {
        v16bf vf = frag_b(vp + ks * 32 + hf * 8);
        oacc[t] = wmma_bf16(pf[ks], vf, oacc[t]);
      }
    }
  }

  // ---- normalize and store attn output bf16 [s][H] ----
#pragma unroll
  for (int r = 0; r < 8; ++r) {
    const float invl = 1.0f / lsum[r];
#pragma unroll
    for (int t = 0; t < 4; ++t) {
      Ob[(size_t)(b * S_ + q0 + hf * 8 + r) * H_ + h * DK_ + t * 16 + lr] =
          (__bf16)(oacc[t][r] * invl);
    }
  }
}

// ---------------- host-side orchestration ----------------
extern "C" void kernel_launch(void* const* d_in, const int* in_sizes, int n_in,
                              void* d_out, int out_size, void* d_ws, size_t ws_size,
                              hipStream_t stream) {
  (void)in_sizes; (void)n_in; (void)out_size; (void)ws_size;
  const float* Qx  = (const float*)d_in[0];
  const float* KVx = (const float*)d_in[1];
  const float* Wq  = (const float*)d_in[2];
  const float* bq  = (const float*)d_in[3];
  const float* Wk  = (const float*)d_in[4];
  const float* bk  = (const float*)d_in[5];
  const float* Wv  = (const float*)d_in[6];
  const float* bv  = (const float*)d_in[7];
  const float* Wo  = (const float*)d_in[8];
  const float* bo  = (const float*)d_in[9];

  const size_t nElem = (size_t)BS_ * H_;  // 4,194,304
  char* base = (char*)d_ws;
  size_t off = 0;
  auto take = [&](size_t bytes) -> void* {
    void* p = base + off;
    off = (off + bytes + 255) & ~(size_t)255;
    return p;
  };
  __bf16* xq  = (__bf16*)take(nElem * 2);
  __bf16* xkv = (__bf16*)take(nElem * 2);
  __bf16* wqT = (__bf16*)take((size_t)H_ * H_ * 2);
  __bf16* wkT = (__bf16*)take((size_t)H_ * H_ * 2);
  __bf16* wvT = (__bf16*)take((size_t)H_ * H_ * 2);
  __bf16* woT = (__bf16*)take((size_t)H_ * H_ * 2);
  __bf16* qb  = (__bf16*)take(nElem * 2);
  __bf16* kb  = (__bf16*)take(nElem * 2);
  __bf16* vtb = (__bf16*)take(nElem * 2);  // [H][BS] transposed V
  __bf16* ob  = (__bf16*)take(nElem * 2);

  // converts
  mha_cvt_bf16<<<(unsigned)(nElem / 1024), 256, 0, stream>>>(Qx, xq, (int)nElem);
  mha_cvt_bf16<<<(unsigned)(nElem / 1024), 256, 0, stream>>>(KVx, xkv, (int)nElem);
  mha_wt_bf16<<<(H_ * H_) / 256, 256, 0, stream>>>(Wq, wqT);
  mha_wt_bf16<<<(H_ * H_) / 256, 256, 0, stream>>>(Wk, wkT);
  mha_wt_bf16<<<(H_ * H_) / 256, 256, 0, stream>>>(Wv, wvT);
  mha_wt_bf16<<<(H_ * H_) / 256, 256, 0, stream>>>(Wo, woT);

  // projections (Q pre-scaled by 1/sqrt(dk) = 0.125, bias included in scale)
  dim3 gg(BS_ / 128, H_ / 128);
  mha_gemm<0><<<gg, 256, 0, stream>>>(xq,  wqT, bq, qb,  BS_, H_, H_, 0.125f);
  mha_gemm<0><<<gg, 256, 0, stream>>>(xkv, wkT, bk, kb,  BS_, H_, H_, 1.0f);
  mha_gemm<1><<<gg, 256, 0, stream>>>(xkv, wvT, bv, vtb, BS_, H_, H_, 1.0f);

  // attention
  dim3 ga(S_ / 64, NH_, B_);
  mha_flash<<<ga, 128, 0, stream>>>(qb, kb, vtb, ob);

  // output projection (f32 out)
  mha_gemm<2><<<gg, 256, 0, stream>>>(ob, woT, bo, d_out, BS_, H_, H_, 1.0f);
}
